// attention_preference_fov_prediction_44117904065311
// MI455X (gfx1250) — matrix operand order repaired and checked by reading, not compiled, not run on hardware
//
#include <hip/hip_runtime.h>
#include <hip/hip_fp16.h>
#include <math.h>

typedef __attribute__((ext_vector_type(16))) _Float16 v16h;
typedef __attribute__((ext_vector_type(8)))  _Float16 v8h;
typedef __attribute__((ext_vector_type(8)))  float    v8f;
typedef __attribute__((ext_vector_type(4)))  float    v4f;

#define WMMA_F16(a,b,c) __builtin_amdgcn_wmma_f32_16x16x32_f16(false,(a),false,(b),(short)0,(c),false,false)

__device__ inline v8f zero8() {
    v8f z;
#pragma unroll
    for (int i = 0; i < 8; ++i) z[i] = 0.f;
    return z;
}

// Load 16 consecutive f32 starting at p (16B aligned) and convert to 16 halfs.
__device__ inline v16h loadcvt16(const float* p) {
    v4f x0 = *(const v4f*)(p);
    v4f x1 = *(const v4f*)(p + 4);
    v4f x2 = *(const v4f*)(p + 8);
    v4f x3 = *(const v4f*)(p + 12);
    v16h b;
#pragma unroll
    for (int i = 0; i < 4; ++i) {
        b[i]      = (_Float16)x0[i];
        b[4 + i]  = (_Float16)x1[i];
        b[8 + i]  = (_Float16)x2[i];
        b[12 + i] = (_Float16)x3[i];
    }
    return b;
}

// B-matrix gather: 8 f32 at row + base and 8 at row + base + 16; each zero if >= limit.
// limit is a multiple of 16, so validity is uniform over each 8-run.
__device__ inline v16h loadcvt16_guard(const float* row, int base, int limit) {
    v16h z;
#pragma unroll
    for (int h = 0; h < 2; ++h) {
        const int bb = base + h * 16;
        if (bb < limit) {
            v4f x0 = *(const v4f*)(row + bb);
            v4f x1 = *(const v4f*)(row + bb + 4);
#pragma unroll
            for (int i = 0; i < 4; ++i) { z[h * 8 + i] = (_Float16)x0[i]; z[h * 8 + 4 + i] = (_Float16)x1[i]; }
        } else {
#pragma unroll
            for (int i = 0; i < 8; ++i) z[h * 8 + i] = (_Float16)0.f;
        }
    }
    return z;
}

// A-matrix pack (16-bit A 16x32 per ISA layout): per-lane halfs
//   a[j] = A[m][ kb + (j/8)*16 + lh*8 + (j%8) ]
__device__ inline v16h packA_f32(const float* prow, int kb, int lh, int limit) {
    v16h a;
    const int b0 = kb + lh * 8;
    const int b1 = kb + 16 + lh * 8;
    if (b0 < limit) {
        v4f x0 = *(const v4f*)(prow + b0);
        v4f x1 = *(const v4f*)(prow + b0 + 4);
#pragma unroll
        for (int i = 0; i < 4; ++i) { a[i] = (_Float16)x0[i]; a[4 + i] = (_Float16)x1[i]; }
    } else {
#pragma unroll
        for (int i = 0; i < 8; ++i) a[i] = (_Float16)0.f;
    }
    if (b1 < limit) {
        v4f x0 = *(const v4f*)(prow + b1);
        v4f x1 = *(const v4f*)(prow + b1 + 4);
#pragma unroll
        for (int i = 0; i < 4; ++i) { a[8 + i] = (_Float16)x0[i]; a[12 + i] = (_Float16)x1[i]; }
    } else {
#pragma unroll
        for (int i = 0; i < 8; ++i) a[8 + i] = (_Float16)0.f;
    }
    return a;
}

// ---------------------------------------------------------------- k_init
__global__ __launch_bounds__(256) void k_init(float* pref, float* hstate) {
    const int i = blockIdx.x * 256 + threadIdx.x;
    if (i < 256 * 800) pref[i] = 0.f;   // zero padded preference buffer
    if (i < 16 * 800) hstate[i] = 0.f;  // zero h0 (f32 carry state)
}

// ---------------------------------------------------------------- weight swizzle
// Pre-swizzle Wv_w into per-lane f16 WMMA B fragments: frag = ks*8 + nt.
__global__ __launch_bounds__(32) void k_wv_swz(const float* __restrict__ wv,
                                               _Float16* __restrict__ wvs) {
    const int frag = blockIdx.x;            // 0..135  (ks in [0,17), nt in [0,8))
    const int ks = frag / 8, nt = frag % 8;
    const int lane = threadIdx.x & 31;
    const int l16 = lane & 15, lh = lane >> 4;
    v16h b = loadcvt16_guard(wv + (size_t)(nt * 16 + l16) * 528, ks * 32 + lh * 8, 528);
    *(v16h*)(wvs + ((size_t)frag * 32 + lane) * 16) = b;
}

// Pre-swizzle gru_w_hh (2352x784) into f16 B fragments: frag = nt*25 + ks,
// nt = g*49 + j over the 2352-wide gate dimension, K padded 784->800 with zeros.
__global__ __launch_bounds__(32) void k_whh_swz(const float* __restrict__ whh,
                                                _Float16* __restrict__ whhs) {
    const int frag = blockIdx.x;            // 0..3674 (nt in [0,147), ks in [0,25))
    const int nt = frag / 25, ks = frag % 25;
    const int lane = threadIdx.x & 31;
    const int l16 = lane & 15, lh = lane >> 4;
    const int col = nt * 16 + l16;
    v16h b = loadcvt16_guard(whh + (size_t)col * 784, ks * 32 + lh * 8, 784);
    *(v16h*)(whhs + ((size_t)frag * 32 + lane) * 16) = b;
}

// ---------------------------------------------------------------- k_gi
// gi_all[256][2352] = fov[256][528] @ gru_w_ih^T + b_ih   (WMMA f16, f32 acc)
__global__ __launch_bounds__(256) void k_gi(const float* __restrict__ fov,
                                            const float* __restrict__ wih,
                                            const float* __restrict__ bih,
                                            float* __restrict__ gi) {
    const int lane = threadIdx.x & 31, wid = threadIdx.x >> 5;
    const int l16 = lane & 15, lh = lane >> 4;
    const int mt = blockIdx.x;              // 0..15
    const int nt = blockIdx.y * 8 + wid;    // 0..146
    if (nt >= 147) return;                  // wave-uniform
    const int row = mt * 16 + l16;
    const int col = nt * 16 + l16;
    const float* pa = fov + (size_t)row * 528;
    const float* pb = wih + (size_t)col * 528;
    v8f acc = zero8();
    for (int ks = 0; ks < 17; ++ks) {       // K = 528 padded to 544
        const int kb = ks * 32;
        v16h a = packA_f32(pa, kb, lh, 528);
        v16h b = loadcvt16_guard(pb, kb + lh * 8, 528);
        acc = WMMA_F16(a, b, acc);
    }
    const float bcol = bih[col];
#pragma unroll
    for (int v = 0; v < 8; ++v) {
        const int r = mt * 16 + v + 8 * lh;
        gi[(size_t)r * 2352 + col] = acc[v] + bcol;
    }
}

// ---------------------------------------------------------------- k_gru
// Single-WGP recurrent kernel: 32 waves, 32 time steps.
// gh = h @ W_hh^T via WMMA with pre-swizzled f16 B fragments (L2-resident).
__global__ __launch_bounds__(1024) void k_gru(const _Float16* __restrict__ whhs,
                                              const float* __restrict__ bhh,
                                              const float* __restrict__ gi,
                                              float* __restrict__ pref,
                                              float* __restrict__ hstate) {
    __shared__ alignas(16) _Float16 h16[2][16][800];
    for (int i = threadIdx.x; i < 2 * 16 * 800; i += 1024)
        ((_Float16*)h16)[i] = (_Float16)0.f;
    __syncthreads();

    const int lane = threadIdx.x & 31, wid = threadIdx.x >> 5;
    const int l16 = lane & 15, lh = lane >> 4;

    for (int s = 0; s < 32; ++s) {
        const int cur = s & 1, nxt = cur ^ 1;
        for (int j = wid; j < 49; j += 32) {
            const int colr = j * 16 + l16;  // h column 0..783
            // pre-swizzled fragment pointers, gate tiles j, j+49, j+98
            const _Float16* f_r = whhs + ((size_t)((j)       * 25) * 32 + lane) * 16;
            const _Float16* f_z = whhs + ((size_t)((49 + j)  * 25) * 32 + lane) * 16;
            const _Float16* f_n = whhs + ((size_t)((98 + j)  * 25) * 32 + lane) * 16;
            v8f ar = zero8(), az = zero8(), an = zero8();
            for (int ks = 0; ks < 25; ++ks) {   // K = 784 pad 800 (pad zero)
                const int kb = ks * 32;
                v8h g0 = *(const v8h*)&h16[cur][l16][kb + lh * 8];
                v8h g1 = *(const v8h*)&h16[cur][l16][kb + 16 + lh * 8];
                v16h a = __builtin_shufflevector(g0, g1, 0, 1, 2, 3, 4, 5, 6, 7,
                                                 8, 9, 10, 11, 12, 13, 14, 15);
                const size_t fo = (size_t)ks * 512;   // 32 lanes * 16 halfs
                v16h br = *(const v16h*)(f_r + fo);
                ar = WMMA_F16(a, br, ar);
                v16h bz = *(const v16h*)(f_z + fo);
                az = WMMA_F16(a, bz, az);
                v16h bn = *(const v16h*)(f_n + fo);
                an = WMMA_F16(a, bn, an);
            }
            if (lh == 0) {  // real batch rows are m = 0..7
#pragma unroll
                for (int v = 0; v < 8; ++v) {
                    const int bb = v;
                    const int row = bb * 32 + s;
                    const float hr = ar[v] + bhh[colr];
                    const float hz = az[v] + bhh[784 + colr];
                    const float hn = an[v] + bhh[1568 + colr];
                    const size_t gbase = (size_t)row * 2352;
                    const float ir = gi[gbase + colr];
                    const float iz = gi[gbase + 784 + colr];
                    const float inn = gi[gbase + 1568 + colr];
                    const float r = 1.f / (1.f + __expf(-(ir + hr)));
                    const float z = 1.f / (1.f + __expf(-(iz + hz)));
                    const float n = tanhf(inn + r * hn);
                    const float hp = hstate[bb * 800 + colr];
                    const float hnew = (1.f - z) * n + z * hp;
                    hstate[bb * 800 + colr] = hnew;     // col-exclusive per wave
                    h16[nxt][bb][colr] = (_Float16)hnew;
                    pref[(size_t)row * 800 + colr] = hnew;
                }
            }
        }
        __syncthreads();
    }
}

// ---------------------------------------------------------------- k_patt
// p_att[256][128] = preference[256][784] @ Wp^T   (WMMA)
__global__ __launch_bounds__(256) void k_patt(const float* __restrict__ pref,
                                              const float* __restrict__ wp,
                                              float* __restrict__ patt) {
    const int lane = threadIdx.x & 31, wid = threadIdx.x >> 5;
    const int l16 = lane & 15, lh = lane >> 4;
    const int mt = blockIdx.x;      // 0..15
    const int nt = wid;             // 0..7
    const int row = mt * 16 + l16;
    const int col = nt * 16 + l16;
    const float* pa = pref + (size_t)row * 800;  // zero-padded to 800
    const float* pb = wp + (size_t)col * 784;
    v8f acc = zero8();
    for (int ks = 0; ks < 25; ++ks) {
        const int kb = ks * 32;
        v16h a = packA_f32(pa, kb, lh, 800);
        v16h b = loadcvt16_guard(pb, kb + lh * 8, 784);
        acc = WMMA_F16(a, b, acc);
    }
#pragma unroll
    for (int v = 0; v < 8; ++v)
        patt[(size_t)(mt * 16 + v + 8 * lh) * 128 + col] = acc[v];
}

// ---------------------------------------------------------------- k_mot
__global__ __launch_bounds__(128) void k_mot(const float* __restrict__ motion,
                                             const float* __restrict__ mew,
                                             const float* __restrict__ meb,
                                             float* __restrict__ motpre) {
    const int r = blockIdx.x, f = threadIdx.x;
    const float* mr = motion + (size_t)r * 90;
    const float* wr = mew + (size_t)f * 90;
    float s = meb[f];
    for (int k = 0; k < 90; ++k) s += mr[k] * wr[k];
    motpre[(size_t)r * 128 + f] = s;
}

// ---------------------------------------------------------------- k_big
// Fused per-t: key GEMM (WMMA) -> tanh/Wpv -> softmax -> att-weighted 1x1 conv.
// video[t] (414KB) is HBM-read once; pass 2 re-reads it from L2.
__global__ __launch_bounds__(256) void k_big(const float* __restrict__ video,
                                             const float* __restrict__ patt,
                                             const _Float16* __restrict__ wvs,
                                             const float* __restrict__ wvb,
                                             const float* __restrict__ wpv,
                                             const float* __restrict__ wpvb,
                                             const float* __restrict__ convw,
                                             const float* __restrict__ convb,
                                             float* __restrict__ vidpre,
                                             float* __restrict__ vidstat) {
    __shared__ float s_cb[128], s_wpv[128];
    __shared__ float s_cw[4 * 528];
    __shared__ float s_sc[208];
    __shared__ float s_att[196];
    __shared__ float s_red[8][8];

    const int t = blockIdx.x, row = t / 5;
    const int tid = threadIdx.x;
    for (int i = tid; i < 128; i += 256) {
        s_cb[i] = wvb[i] + patt[(size_t)row * 128 + i];
        s_wpv[i] = wpv[i];
    }
    for (int i = tid; i < 2112; i += 256) s_cw[i] = convw[i];
    __syncthreads();

    const int lane = tid & 31, wid = tid >> 5, l16 = lane & 15, lh = lane >> 4;
    const float* vt = video + (size_t)t * 103488;
    const _Float16* fb = wvs + (size_t)lane * 16;
    const float bias = wpvb[0];

    // ---- pass 1: scores[196] = Wpv . tanh(key@Wv^T + cb) ----
    for (int mt = wid; mt < 13; mt += 8) {
        const int n0 = mt * 16 + l16;          // spatial position (GEMM M)
        const int n = (n0 < 196) ? n0 : 195;   // clamp: garbage rows discarded below
        const float* va = vt + n;
        v8f acc[8];
#pragma unroll
        for (int q = 0; q < 8; ++q) acc[q] = zero8();
        // 16 full K-steps: channels 0..511, no guards needed anywhere
        for (int ks = 0; ks < 16; ++ks) {
            const int kb = ks * 32;
            const int c0 = kb + lh * 8;
            v16h a;
#pragma unroll
            for (int i = 0; i < 8; ++i)
                a[i] = (_Float16)va[(size_t)(c0 + i) * 196];
#pragma unroll
            for (int i = 0; i < 8; ++i)
                a[8 + i] = (_Float16)va[(size_t)(c0 + 16 + i) * 196];
            if (lane == 0)
                __builtin_prefetch(va + (size_t)(kb + 32) * 196, 0, 1);
#pragma unroll
            for (int nt = 0; nt < 8; ++nt) {  // 128 features, pre-swizzled B
                v16h b = *(const v16h*)(fb + (size_t)(ks * 8 + nt) * 512);
                acc[nt] = WMMA_F16(a, b, acc[nt]);
            }
        }
        {   // peeled K tail: channels 512..527 valid, 528..543 zero (B frag also zero)
            const int c0 = 512 + lh * 8;
            v16h a;
#pragma unroll
            for (int i = 0; i < 8; ++i)
                a[i] = (_Float16)va[(size_t)(c0 + i) * 196];
#pragma unroll
            for (int i = 0; i < 8; ++i)
                a[8 + i] = (_Float16)0.f;
#pragma unroll
            for (int nt = 0; nt < 8; ++nt) {
                v16h b = *(const v16h*)(fb + (size_t)(16 * 8 + nt) * 512);
                acc[nt] = WMMA_F16(a, b, acc[nt]);
            }
        }
        float sv[8];
#pragma unroll
        for (int v = 0; v < 8; ++v) sv[v] = 0.f;
#pragma unroll
        for (int nt = 0; nt < 8; ++nt) {
            const int f = nt * 16 + l16;
            const float wf = s_wpv[f], cbf = s_cb[f];
#pragma unroll
            for (int v = 0; v < 8; ++v) sv[v] += wf * tanhf(acc[nt][v] + cbf);
        }
#pragma unroll
        for (int v = 0; v < 8; ++v) {  // reduce over 16 feature lanes
            float x = sv[v];
            x += __shfl_xor(x, 1);
            x += __shfl_xor(x, 2);
            x += __shfl_xor(x, 4);
            x += __shfl_xor(x, 8);
            sv[v] = x;
        }
        if (l16 == 0) {
#pragma unroll
            for (int v = 0; v < 8; ++v) {
                const int nn = mt * 16 + v + 8 * lh;
                if (nn < 196) s_sc[nn] = sv[v] + bias;
            }
        }
    }
    __syncthreads();

    // ---- softmax over 196 ----
    {
        float x = (tid < 196) ? s_sc[tid] : -3.0e38f;
        float m = x;
#pragma unroll
        for (int d = 16; d >= 1; d >>= 1) m = fmaxf(m, __shfl_xor(m, d));
        if (lane == 0) s_red[0][wid] = m;
        __syncthreads();
        float bm = s_red[0][0];
#pragma unroll
        for (int i = 1; i < 8; ++i) bm = fmaxf(bm, s_red[0][i]);
        float e = (tid < 196) ? __expf(x - bm) : 0.f;
        float sm = e;
#pragma unroll
        for (int d = 16; d >= 1; d >>= 1) sm += __shfl_xor(sm, d);
        if (lane == 0) s_red[1][wid] = sm;
        __syncthreads();
        float tot = 0.f;
#pragma unroll
        for (int i = 0; i < 8; ++i) tot += s_red[1][i];
        if (tid < 196) s_att[tid] = e / tot;
    }
    __syncthreads();

    // ---- pass 2: 1x1 conv over the OTHER view (196,528), scaled by att ----
    float ls[4] = {0.f, 0.f, 0.f, 0.f}, lq[4] = {0.f, 0.f, 0.f, 0.f};
    if (tid < 196) {
        const int n2 = tid;
        const float an = s_att[n2];
        const float* pv = vt + (size_t)n2 * 528;
        float s0 = 0.f, s1 = 0.f, s2 = 0.f, s3 = 0.f;
        for (int c = 0; c < 528; c += 4) {
            v4f vv = *(const v4f*)(pv + c);
#pragma unroll
            for (int i = 0; i < 4; ++i) {
                const float vx = vv[i];
                s0 += vx * s_cw[c + i];
                s1 += vx * s_cw[528 + c + i];
                s2 += vx * s_cw[1056 + c + i];
                s3 += vx * s_cw[1584 + c + i];
            }
        }
        const float o0 = s0 * an + convb[0];
        const float o1 = s1 * an + convb[1];
        const float o2 = s2 * an + convb[2];
        const float o3 = s3 * an + convb[3];
        float* vp = vidpre + (size_t)t * 784;
        vp[0 * 196 + n2] = o0;
        vp[1 * 196 + n2] = o1;
        vp[2 * 196 + n2] = o2;
        vp[3 * 196 + n2] = o3;
        ls[0] = o0; ls[1] = o1; ls[2] = o2; ls[3] = o3;
        lq[0] = o0 * o0; lq[1] = o1 * o1; lq[2] = o2 * o2; lq[3] = o3 * o3;
    }
    // deterministic per-block partial sums for BN stats
#pragma unroll
    for (int o = 0; o < 4; ++o) {
        float x = ls[o];
#pragma unroll
        for (int d = 16; d >= 1; d >>= 1) x += __shfl_xor(x, d);
        if (lane == 0) s_red[wid][o] = x;
        float y = lq[o];
#pragma unroll
        for (int d = 16; d >= 1; d >>= 1) y += __shfl_xor(y, d);
        if (lane == 0) s_red[wid][4 + o] = y;
    }
    __syncthreads();
    if (tid < 8) {
        float a = 0.f;
        for (int w = 0; w < 8; ++w) a += s_red[w][tid];
        vidstat[(size_t)t * 8 + tid] = a;
    }
}

// ---------------------------------------------------------------- k_stats
__global__ __launch_bounds__(256) void k_stats(const float* __restrict__ vidstat,
                                               const float* __restrict__ motpre,
                                               const float* __restrict__ vbng,
                                               const float* __restrict__ vbnb,
                                               const float* __restrict__ mbng,
                                               const float* __restrict__ mbnb,
                                               float* __restrict__ bnp) {
    const int tid = threadIdx.x;
    if (tid < 4) {
        float s = 0.f, q = 0.f;
        for (int t = 0; t < 1280; ++t) {
            s += vidstat[t * 8 + tid];
            q += vidstat[t * 8 + 4 + tid];
        }
        const float N = 1280.f * 196.f;
        const float mu = s / N;
        const float var = q / N - mu * mu;
        const float sc = vbng[tid] * rsqrtf(var + 1e-5f);
        bnp[tid] = sc;
        bnp[4 + tid] = vbnb[tid] - mu * sc;
    }
    if (tid >= 32 && tid < 160) {
        const int f = tid - 32;
        float s = 0.f, q = 0.f;
        for (int r = 0; r < 256; ++r) {
            const float x = motpre[r * 128 + f];
            s += x;
            q += x * x;
        }
        const float mu = s / 256.f;
        const float var = q / 256.f - mu * mu;
        const float sc = mbng[f] * rsqrtf(var + 1e-5f);
        bnp[8 + f] = sc;
        bnp[136 + f] = mbnb[f] - mu * sc;
    }
}

// ---------------------------------------------------------------- k_write
__global__ __launch_bounds__(256) void k_write(const float* __restrict__ pref,
                                               const float* __restrict__ motpre,
                                               const float* __restrict__ vidpre,
                                               const float* __restrict__ bnp,
                                               const float* __restrict__ tw,
                                               float* __restrict__ out) {
    const int t = blockIdx.x, row = t / 5, p = t % 5;
    float* o = out + (size_t)t * 1697;
    for (int j = threadIdx.x; j < 1697; j += 256) {
        float v;
        if (j == 0) {
            v = tw[p];
        } else if (j < 785) {
            v = pref[(size_t)row * 800 + (j - 1)];
        } else if (j < 913) {
            const int f = j - 785;
            v = fmaxf(0.f, motpre[(size_t)row * 128 + f] * bnp[8 + f] + bnp[136 + f]);
        } else {
            const int k = j - 913;
            const int oo = k / 196;
            v = fmaxf(0.f, vidpre[(size_t)t * 784 + k] * bnp[oo] + bnp[4 + oo]);
        }
        o[j] = v;
    }
}

// ---------------------------------------------------------------- launch
extern "C" void kernel_launch(void* const* d_in, const int* in_sizes, int n_in,
                              void* d_out, int out_size, void* d_ws, size_t ws_size,
                              hipStream_t stream) {
    const float* fov    = (const float*)d_in[0];
    const float* motion = (const float*)d_in[1];
    const float* video  = (const float*)d_in[2];
    const float* tw     = (const float*)d_in[3];
    const float* wih    = (const float*)d_in[4];
    const float* whh    = (const float*)d_in[5];
    const float* bih    = (const float*)d_in[6];
    const float* bhh    = (const float*)d_in[7];
    const float* Wp     = (const float*)d_in[8];
    const float* Wv     = (const float*)d_in[9];
    const float* Wvb    = (const float*)d_in[10];
    const float* Wpv    = (const float*)d_in[11];
    const float* Wpvb   = (const float*)d_in[12];
    const float* convw  = (const float*)d_in[13];
    const float* convb  = (const float*)d_in[14];
    const float* vbng   = (const float*)d_in[15];
    const float* vbnb   = (const float*)d_in[16];
    const float* mew    = (const float*)d_in[17];
    const float* meb    = (const float*)d_in[18];
    const float* mbng   = (const float*)d_in[19];
    const float* mbnb   = (const float*)d_in[20];

    float* ws      = (float*)d_ws;
    float* gi      = ws + 0;         // 256*2352  = 602112
    float* pref    = ws + 602112;    // 256*800   = 204800 (H=784 padded)
    float* patt    = ws + 806912;    // 256*128   = 32768
    float* vidpre  = ws + 839680;    // 1280*784  = 1003520
    float* vidstat = ws + 1843200;   // 1280*8    = 10240
    float* motpre  = ws + 1853440;   // 256*128   = 32768
    float* bnp     = ws + 1886208;   // 264
    float* hstate  = ws + 1886472;   // 16*800    = 12800
    // f16 regions (32B-aligned float offsets)
    _Float16* whhs = (_Float16*)(ws + 1899272);  // 147*25 frags * 512 halfs = 1,881,600
    _Float16* wvs  = (_Float16*)(ws + 2840072);  // 17*8 frags * 512 halfs   = 69,632
    float* out     = (float*)d_out;

    k_init<<<850, 256, 0, stream>>>(pref, hstate);
    k_wv_swz<<<136, 32, 0, stream>>>(Wv, wvs);
    k_whh_swz<<<3675, 32, 0, stream>>>(whh, whhs);
    k_gi<<<dim3(16, 19), 256, 0, stream>>>(fov, wih, bih, gi);
    k_gru<<<1, 1024, 0, stream>>>(whhs, bhh, gi, pref, hstate);
    k_patt<<<16, 256, 0, stream>>>(pref, Wp, patt);
    k_mot<<<256, 128, 0, stream>>>(motion, mew, meb, motpre);
    k_big<<<1280, 256, 0, stream>>>(video, patt, wvs, Wvb, Wpv, Wpvb,
                                    convw, convb, vidpre, vidstat);
    k_stats<<<1, 256, 0, stream>>>(vidstat, motpre, vbng, vbnb, mbng, mbnb, bnp);
    k_write<<<1280, 256, 0, stream>>>(pref, motpre, vidpre, bnp, tw, out);
}
